// MultiFactorAttention_31739808317782
// MI455X (gfx1250) — matrix operand order repaired and checked by reading, not compiled, not run on hardware
//
#include <hip/hip_runtime.h>

typedef __attribute__((ext_vector_type(16))) _Float16 v16h;
typedef __attribute__((ext_vector_type(8)))  _Float16 v8h;
typedef __attribute__((ext_vector_type(8)))  float    v8f;
typedef __attribute__((ext_vector_type(4)))  float    v4f;

#define B_  64
#define T_  1024
#define S_  1536
#define D_  512
#define F_  32
#define INV_SCALE 0.04419417382415922f  /* 1/sqrt(512) */

static __device__ __forceinline__ int laneid() { return (int)(threadIdx.x & 31); }

static __device__ __forceinline__ v8f wmma16(v16h a, v16h b, v8f c) {
  // D = A(16x32 f16) * B(32x16 f16) + C(16x16 f32)
  return __builtin_amdgcn_wmma_f32_16x16x32_f16(false, a, false, b, (short)0, c, false, false);
}

// ---------- fragment loaders (wave32 CDNA5 WMMA layouts) ----------

// A fragment (16x32) from row-major f32 source, vectorized (4x float4, 32B aligned).
// lane: M = lane&15 ; K = koff + h (+8 for h>=8), koff = 8*(lane>=16)
static __device__ __forceinline__ v16h load_a_f32vec(const float* __restrict__ p, int ld,
                                                     int row0, int col0) {
  const int l = laneid();
  const int m = l & 15;
  const int koff = (l >> 4) << 3;
  const float* rp = p + (size_t)(row0 + m) * ld + col0 + koff;
  const v4f f0 = *(const v4f*)(rp);        // K = koff+0 ..3
  const v4f f1 = *(const v4f*)(rp + 4);    // K = koff+4 ..7
  const v4f f2 = *(const v4f*)(rp + 16);   // K = koff+16..19
  const v4f f3 = *(const v4f*)(rp + 20);   // K = koff+20..23
  v16h a;
#pragma unroll
  for (int h = 0; h < 4; ++h) {
    a[h]      = (_Float16)f0[h];
    a[h + 4]  = (_Float16)f1[h];
    a[h + 8]  = (_Float16)f2[h];
    a[h + 12] = (_Float16)f3[h];
  }
  return a;
}

// A fragment (16x32) from row-major f16 source (two contiguous 8-half runs).
static __device__ __forceinline__ v16h load_a_f16src(const _Float16* __restrict__ p, int ld,
                                                     int row0, int col0) {
  const int l = laneid();
  const int m = l & 15;
  const int koff = (l >> 4) << 3;
  const _Float16* rp = p + (size_t)(row0 + m) * ld + col0 + koff;
  const v8h lo = *(const v8h*)(rp);        // K = koff .. koff+7
  const v8h hi = *(const v8h*)(rp + 16);   // K = koff+16 .. koff+23
  v16h a;
#pragma unroll
  for (int h = 0; h < 8; ++h) { a[h] = lo[h]; a[h + 8] = hi[h]; }
  return a;
}

// B fragment (32x16): B(k,n) = M[n0+n][k0+k] with M row-major f16 ([N x K], i.e.
// a transposed weight, V^T, or K for Q.K^T). two 8-half runs per lane, 16 apart (2x16B).
static __device__ __forceinline__ v16h load_bt_f16src(const _Float16* __restrict__ p, int ld,
                                                      int n0, int k0) {
  const int l = laneid();
  const int n = l & 15;
  const int kb = (l >> 4) << 3;
  const _Float16* rp = p + (size_t)(n0 + n) * ld + k0 + kb;
  const v8h lo = *(const v8h*)(rp);
  const v8h hi = *(const v8h*)(rp + 16);
  v16h b;
#pragma unroll
  for (int h = 0; h < 8; ++h) { b[h] = lo[h]; b[h + 8] = hi[h]; }
  return b;
}

// D store (16x16 f32 acc) -> row-major f16: lane: N = lane&15 ; M = mo + r.
static __device__ __forceinline__ void store_d_f16(_Float16* __restrict__ p, int ld,
                                                   int row0, int col0, v8f c) {
  const int l = laneid();
  const int n = l & 15;
  const int mo = (l >> 4) << 3;
  _Float16* wp = p + (size_t)(row0 + mo) * ld + col0 + n;
#pragma unroll
  for (int r = 0; r < 8; ++r) wp[(size_t)r * ld] = (_Float16)c[r];
}

static __device__ __forceinline__ void store_d_f16_bias(_Float16* __restrict__ p, int ld,
                                                        int row0, int col0, v8f c,
                                                        const float* __restrict__ bias) {
  const int l = laneid();
  const int n = l & 15;
  const int mo = (l >> 4) << 3;
  const float bv = bias[col0 + n];
  _Float16* wp = p + (size_t)(row0 + mo) * ld + col0 + n;
#pragma unroll
  for (int r = 0; r < 8; ++r) wp[(size_t)r * ld] = (_Float16)(c[r] + bv);
}

// D store transposed -> dst[N x M] row-major f16: per lane the 8 row values are
// CONTIGUOUS in memory -> single aligned 16B store. Used to build V^T.
static __device__ __forceinline__ void store_dT_f16_bias(_Float16* __restrict__ p, int ldT,
                                                         int row0, int col0, v8f c,
                                                         const float* __restrict__ bias) {
  const int l = laneid();
  const int n = l & 15;
  const int mo = (l >> 4) << 3;
  const float bv = bias[col0 + n];
  v8h v;
#pragma unroll
  for (int r = 0; r < 8; ++r) v[r] = (_Float16)(c[r] + bv);
  *(v8h*)(p + (size_t)(col0 + n) * ldT + row0 + mo) = v;
}

// ------------- kernel 0: one-time weight transpose + f32->f16 convert -------------
// dst[n*K + k] = (f16) src[k*N + n]   (src is [K x N] row-major)
__global__ __launch_bounds__(256) void wtrans_kernel(const float* __restrict__ src,
                                                     _Float16* __restrict__ dst,
                                                     int K, int N) {
  const int idx = blockIdx.x * 256 + threadIdx.x;
  if (idx < K * N) {
    const int k = idx / N, n = idx % N;
    dst[(size_t)n * K + k] = (_Float16)src[idx];
  }
}

// ---------------- kernel 1: K/V projection (K-dim = F = 32, one WMMA each) ----------
// K stored row-major [S x D] f16 (read later as contiguous K^T fragments).
// V stored transposed [D x S] f16 (contiguous B fragments + contiguous b128 stores).
__global__ __launch_bounds__(256) void kv_proj_kernel(
    const float* __restrict__ wh, const float* __restrict__ wf,
    const _Float16* __restrict__ Wkt, const float* __restrict__ bk,
    const _Float16* __restrict__ Wvt, const float* __restrict__ bv,
    _Float16* __restrict__ Kh, _Float16* __restrict__ Vt) {
  const int gid = blockIdx.x * 8 + (threadIdx.x >> 5);
  const int dt = gid & 31;                 // D/16 = 32 tiles
  const int st = (gid >> 5) % 96;          // S/16 = 96 tiles
  const int b  = gid / (32 * 96);
  const int s0 = st * 16, d0 = dt * 16;

  // A fragment: weather rows (concat handled by address select), vectorized f32.
  const int l = laneid();
  const int m = l & 15;
  const int koff = (l >> 4) << 3;
  const int s = s0 + m;
  const float* rp = ((s < T_) ? (wh + ((size_t)b * T_ + s) * F_)
                              : (wf + ((size_t)b * (S_ - T_) + (s - T_)) * F_)) + koff;
  const v4f f0 = *(const v4f*)(rp);
  const v4f f1 = *(const v4f*)(rp + 4);
  const v4f f2 = *(const v4f*)(rp + 16);
  const v4f f3 = *(const v4f*)(rp + 20);
  v16h a;
#pragma unroll
  for (int h = 0; h < 4; ++h) {
    a[h] = (_Float16)f0[h]; a[h + 4] = (_Float16)f1[h];
    a[h + 8] = (_Float16)f2[h]; a[h + 12] = (_Float16)f3[h];
  }

  v8f z = {};
  v8f ck = wmma16(a, load_bt_f16src(Wkt, F_, d0, 0), z);
  v8f cv = wmma16(a, load_bt_f16src(Wvt, F_, d0, 0), z);
  store_d_f16_bias(Kh + (size_t)b * S_ * D_, D_, s0, d0, ck, bk);
  store_dT_f16_bias(Vt + (size_t)b * D_ * S_, S_, s0, d0, cv, bv);
}

// ---------------- kernel 2: Q projection (one wave -> 4 D-tiles, A reused) ----------
__global__ __launch_bounds__(256) void q_proj_kernel(
    const float* __restrict__ y, const _Float16* __restrict__ Wqt,
    const float* __restrict__ bq, _Float16* __restrict__ Qh) {
  const int gid = blockIdx.x * 8 + (threadIdx.x >> 5);
  const int g  = gid & 7;                  // 8 groups of 4 D-tiles
  const int tt = (gid >> 3) & 63;
  const int b  = gid >> 9;
  const int t0 = tt * 16;
  const float* yb = y + (size_t)b * T_ * D_;

  v16h aq[16];
#pragma unroll
  for (int kk = 0; kk < 16; ++kk) aq[kk] = load_a_f32vec(yb, D_, t0, kk * 32);

  _Float16* Qb = Qh + (size_t)b * T_ * D_;
#pragma unroll
  for (int i = 0; i < 4; ++i) {
    const int d0 = (g * 4 + i) * 16;
    v8f c = {};
#pragma unroll
    for (int kk = 0; kk < 16; ++kk)
      c = wmma16(aq[kk], load_bt_f16src(Wqt, D_, d0, kk * 32), c);
    store_d_f16_bias(Qb, D_, t0, d0, c, bq);
  }
}

// ------- kernel 3: attention (scores + softmax in LDS + attn@V), 8 waves/block -------
__global__ __launch_bounds__(256) void attn_kernel(
    const _Float16* __restrict__ Qh, const _Float16* __restrict__ Kh,
    const _Float16* __restrict__ Vt, _Float16* __restrict__ Ch) {
  __shared__ _Float16 sP[16 * S_];   // 48 KB: 16 T-rows x 1536 S-cols
  __shared__ float red[16][16];
  __shared__ float rowstat[16];

  const int b   = blockIdx.x >> 6;
  const int t0  = (blockIdx.x & 63) * 16;
  const int tid = threadIdx.x;
  const int wv  = tid >> 5;

  const _Float16* Qb = Qh + (size_t)b * T_ * D_;
  const _Float16* Kb = Kh + (size_t)b * S_ * D_;
  const _Float16* Vb = Vt + (size_t)b * D_ * S_;

  // Preload Q fragments for this 16-row tile (shared across all score chunks)
  v16h aq[16];
#pragma unroll
  for (int kk = 0; kk < 16; ++kk) aq[kk] = load_a_f16src(Qb, D_, t0, kk * 32);

  const int l  = laneid();
  const int n  = l & 15;
  const int mo = (l >> 4) << 3;

  // ---- scores: each wave handles 12 of the 96 S-chunks ----
  for (int j = wv; j < S_ / 16; j += 8) {
    const int s0 = j * 16;
    v8f c = {};
#pragma unroll
    for (int kk = 0; kk < 16; ++kk)
      c = wmma16(aq[kk], load_bt_f16src(Kb, D_, s0, kk * 32), c);
#pragma unroll
    for (int r = 0; r < 8; ++r)
      sP[(mo + r) * S_ + s0 + n] = (_Float16)(c[r] * INV_SCALE);
  }
  __syncthreads();

  // ---- softmax over S per row (16 threads per row) ----
  const int row = tid >> 4;
  const int sub = tid & 15;
  float mx = -1e30f;
  for (int j = sub; j < S_; j += 16) mx = fmaxf(mx, (float)sP[row * S_ + j]);
  red[row][sub] = mx;
  __syncthreads();
  if (sub == 0) {
    float m2 = red[row][0];
#pragma unroll
    for (int i = 1; i < 16; ++i) m2 = fmaxf(m2, red[row][i]);
    rowstat[row] = m2;
  }
  __syncthreads();
  const float rmax = rowstat[row];
  float sm = 0.f;
  for (int j = sub; j < S_; j += 16) {
    const float e = __expf((float)sP[row * S_ + j] - rmax);
    sP[row * S_ + j] = (_Float16)e;
    sm += e;
  }
  red[row][sub] = sm;
  __syncthreads();
  if (sub == 0) {
    float s2 = 0.f;
#pragma unroll
    for (int i = 0; i < 16; ++i) s2 += red[row][i];
    rowstat[row] = s2;
  }
  __syncthreads();
  const float rinv = 1.0f / rowstat[row];
  for (int j = sub; j < S_; j += 16)
    sP[row * S_ + j] = (_Float16)((float)sP[row * S_ + j] * rinv);
  __syncthreads();

  // ---- attn @ V : each wave owns a 64-wide D slice (4 output tiles).
  // B fragments come from V^T -> fully contiguous 2x16B loads per lane.
  v8f a0 = {}, a1 = {}, a2 = {}, a3 = {};
  const int d0 = wv * 64;
  for (int sc = 0; sc < S_ / 32; ++sc) {
    const v16h af = load_a_f16src(sP, S_, 0, sc * 32);   // attn frag from LDS
    a0 = wmma16(af, load_bt_f16src(Vb, S_, d0 +  0, sc * 32), a0);
    a1 = wmma16(af, load_bt_f16src(Vb, S_, d0 + 16, sc * 32), a1);
    a2 = wmma16(af, load_bt_f16src(Vb, S_, d0 + 32, sc * 32), a2);
    a3 = wmma16(af, load_bt_f16src(Vb, S_, d0 + 48, sc * 32), a3);
  }
  _Float16* Cb = Ch + (size_t)b * T_ * D_;
  store_d_f16(Cb, D_, t0, d0 +  0, a0);
  store_d_f16(Cb, D_, t0, d0 + 16, a1);
  store_d_f16(Cb, D_, t0, d0 + 32, a2);
  store_d_f16(Cb, D_, t0, d0 + 48, a3);
}

// ----- kernel 4: logits[b,t] = ((corr @ Ws + bs) . y_hat) / SCALE (fused, WMMA) -----
__global__ __launch_bounds__(256) void gate_logits_kernel(
    const _Float16* __restrict__ Ch, const _Float16* __restrict__ Wst,
    const float* __restrict__ bs, const float* __restrict__ y,
    float* __restrict__ logits) {
  __shared__ float rowsum[16];
  const int b   = blockIdx.x >> 6;
  const int t0  = (blockIdx.x & 63) * 16;
  const int tid = threadIdx.x;
  const int wv  = tid >> 5;
  if (tid < 16) rowsum[tid] = 0.f;
  __syncthreads();

  const _Float16* Cb = Ch + (size_t)b * T_ * D_;
  v16h aq[16];
#pragma unroll
  for (int kk = 0; kk < 16; ++kk) aq[kk] = load_a_f16src(Cb, D_, t0, kk * 32);

  const int l  = laneid();
  const int n  = l & 15;
  const int mo = (l >> 4) << 3;
#pragma unroll
  for (int i = 0; i < 4; ++i) {
    const int d0 = (wv + i * 8) * 16;      // wave-owned D tiles
    v8f c = {};
#pragma unroll
    for (int kk = 0; kk < 16; ++kk)
      c = wmma16(aq[kk], load_bt_f16src(Wst, D_, d0, kk * 32), c);
    const float bv = bs[d0 + n];
    const float* yp = y + ((size_t)b * T_ + t0 + mo) * D_ + d0 + n;
#pragma unroll
    for (int r = 0; r < 8; ++r)
      atomicAdd(&rowsum[mo + r], (c[r] + bv) * yp[(size_t)r * D_]);
  }
  __syncthreads();
  if (tid < 16) logits[(size_t)b * T_ + t0 + tid] = rowsum[tid] * INV_SCALE;
}

// ---------------- kernel 5a: softmax over T + pooled final vector ----------------
__global__ __launch_bounds__(256) void pool_kernel(
    const float* __restrict__ logits, const float* __restrict__ y,
    float* __restrict__ finals) {
  __shared__ float p[T_];
  __shared__ float red[256];
  const int b   = blockIdx.x;
  const int tid = threadIdx.x;
  const float* lg = logits + (size_t)b * T_;

  float mx = -1e30f;
  for (int t = tid; t < T_; t += 256) mx = fmaxf(mx, lg[t]);
  red[tid] = mx;
  __syncthreads();
  for (int off = 128; off > 0; off >>= 1) {
    if (tid < off) red[tid] = fmaxf(red[tid], red[tid + off]);
    __syncthreads();
  }
  const float bmax = red[0];
  __syncthreads();

  float sm = 0.f;
  for (int t = tid; t < T_; t += 256) { const float e = __expf(lg[t] - bmax); p[t] = e; sm += e; }
  red[tid] = sm;
  __syncthreads();
  for (int off = 128; off > 0; off >>= 1) {
    if (tid < off) red[tid] += red[tid + off];
    __syncthreads();
  }
  const float inv = 1.0f / red[0];
  __syncthreads();

  for (int d = tid; d < D_; d += 256) {
    float acc = 0.f;
    const float* yp = y + (size_t)b * T_ * D_ + d;
    for (int t = 0; t < T_; ++t) acc += p[t] * yp[(size_t)t * D_];
    finals[(size_t)b * D_ + d] = acc * inv;
  }
}

// ---------------- kernel 5b: out = y_hat + broadcast(final) ----------------
__global__ __launch_bounds__(256) void add_final_kernel(
    const float* __restrict__ y, const float* __restrict__ finals,
    float* __restrict__ out, size_t N) {
  size_t i = (size_t)blockIdx.x * blockDim.x + threadIdx.x;
  const size_t stride = (size_t)gridDim.x * blockDim.x;
  for (; i < N; i += stride) {
    const size_t b = i / ((size_t)T_ * D_);
    const size_t d = i % D_;
    out[i] = y[i] + finals[b * D_ + d];
  }
}

extern "C" void kernel_launch(void* const* d_in, const int* in_sizes, int n_in,
                              void* d_out, int out_size, void* d_ws, size_t ws_size,
                              hipStream_t stream) {
  const float* y  = (const float*)d_in[0];
  const float* wh = (const float*)d_in[1];
  const float* wf = (const float*)d_in[2];
  const float* Wq = (const float*)d_in[3];
  const float* bq = (const float*)d_in[4];
  const float* Wk = (const float*)d_in[5];
  const float* bk = (const float*)d_in[6];
  const float* Wv = (const float*)d_in[7];
  const float* bv = (const float*)d_in[8];
  const float* Ws = (const float*)d_in[9];
  const float* bs = (const float*)d_in[10];
  float* out = (float*)d_out;

  // workspace layout (f16 staging halves HBM traffic and feeds WMMA directly)
  char* w = (char*)d_ws;
  _Float16* Qh  = (_Float16*)w; w += (size_t)B_ * T_ * D_ * sizeof(_Float16);
  _Float16* Kh  = (_Float16*)w; w += (size_t)B_ * S_ * D_ * sizeof(_Float16);
  _Float16* Vt  = (_Float16*)w; w += (size_t)B_ * D_ * S_ * sizeof(_Float16);
  _Float16* Ch  = (_Float16*)w; w += (size_t)B_ * T_ * D_ * sizeof(_Float16);
  float* logits = (float*)w;    w += (size_t)B_ * T_ * sizeof(float);
  float* finals = (float*)w;    w += (size_t)B_ * D_ * sizeof(float);
  _Float16* Wqt = (_Float16*)w; w += (size_t)D_ * D_ * sizeof(_Float16);
  _Float16* Wst = (_Float16*)w; w += (size_t)D_ * D_ * sizeof(_Float16);
  _Float16* Wkt = (_Float16*)w; w += (size_t)F_ * D_ * sizeof(_Float16);
  _Float16* Wvt = (_Float16*)w; w += (size_t)F_ * D_ * sizeof(_Float16);

  // one-time weight transposes (f32 [KxN] -> f16 [NxK])
  wtrans_kernel<<<(D_ * D_ + 255) / 256, 256, 0, stream>>>(Wq, Wqt, D_, D_);
  wtrans_kernel<<<(D_ * D_ + 255) / 256, 256, 0, stream>>>(Ws, Wst, D_, D_);
  wtrans_kernel<<<(F_ * D_ + 255) / 256, 256, 0, stream>>>(Wk, Wkt, F_, D_);
  wtrans_kernel<<<(F_ * D_ + 255) / 256, 256, 0, stream>>>(Wv, Wvt, F_, D_);

  kv_proj_kernel<<<(B_ * (S_ / 16) * (D_ / 16)) / 8, 256, 0, stream>>>(
      wh, wf, Wkt, bk, Wvt, bv, Kh, Vt);
  q_proj_kernel<<<(B_ * (T_ / 16) * (D_ / 16) / 4) / 8, 256, 0, stream>>>(y, Wqt, bq, Qh);
  attn_kernel<<<B_ * (T_ / 16), 256, 0, stream>>>(Qh, Kh, Vt, Ch);
  gate_logits_kernel<<<B_ * (T_ / 16), 256, 0, stream>>>(Ch, Wst, bs, y, logits);
  pool_kernel<<<B_, 256, 0, stream>>>(logits, y, finals);
  const size_t N = (size_t)B_ * T_ * D_;
  add_final_kernel<<<32768, 256, 0, stream>>>(y, finals, out, N);
}
